// SpikingNeuralNetwork_42442866819833
// MI455X (gfx1250) — matrix operand order repaired and checked by reading, not compiled, not run on hardware
//
#include <hip/hip_runtime.h>

typedef __attribute__((ext_vector_type(4)))  float  v4f;
typedef __attribute__((ext_vector_type(8)))  float  v8f;
typedef __attribute__((ext_vector_type(4)))  __bf16 v4bf;
typedef __attribute__((ext_vector_type(8)))  __bf16 v8bf;
typedef __attribute__((ext_vector_type(16))) __bf16 v16bf;

#define KC  64      // K chunk staged per buffer (2 wmma K-steps)
#define STR 72      // LDS row stride in bf16 (144B: 16B-aligned, bank-conflict-free)

// ---------------------------------------------------------------------------
// bf16 helpers (RNE via bit ops)
// ---------------------------------------------------------------------------
__device__ inline __bf16 f2bf(float x) {
    unsigned u = __builtin_bit_cast(unsigned, x);
    unsigned r = (u + 0x7FFFu + ((u >> 16) & 1u)) >> 16;
    return __builtin_bit_cast(__bf16, (unsigned short)r);
}
__device__ inline float bf2f(__bf16 b) {
    unsigned u = ((unsigned)__builtin_bit_cast(unsigned short, b)) << 16;
    return __builtin_bit_cast(float, u);
}

// ---------------------------------------------------------------------------
// CDNA5 async global->LDS copy (per-lane 16B), tracked by ASYNCcnt
// ---------------------------------------------------------------------------
__device__ inline void async_b128(unsigned lds_off_bytes, const void* gptr) {
    asm volatile("global_load_async_to_lds_b128 %0, %1, off"
                 :: "v"(lds_off_bytes), "v"(gptr) : "memory");
}
template <int N>
__device__ inline void wait_async_le() {
    asm volatile("s_wait_asynccnt %0" :: "n"(N) : "memory");
}

// ---------------------------------------------------------------------------
// Zero-fill
// ---------------------------------------------------------------------------
__global__ void snn_zero_kernel(float* __restrict__ p, int n4) {
    int i = blockIdx.x * blockDim.x + threadIdx.x;
    if (i < n4) {
        v4f z = {0.0f, 0.0f, 0.0f, 0.0f};
        ((v4f*)p)[i] = z;
    }
}

// ---------------------------------------------------------------------------
// f32 -> (bf16 hi, bf16 lo) split:  hi = bf16(x), lo = bf16(x - hi)
// ---------------------------------------------------------------------------
__global__ void snn_split_kernel(const float* __restrict__ src,
                                 __bf16* __restrict__ hi,
                                 __bf16* __restrict__ lo, int n4) {
    int i = blockIdx.x * blockDim.x + threadIdx.x;
    if (i < n4) {
        v4f s = ((const v4f*)src)[i];
        v4bf h, l;
#pragma unroll
        for (int j = 0; j < 4; j++) {
            __bf16 hb = f2bf(s[j]);
            h[j] = hb;
            l[j] = f2bf(s[j] - bf2f(hb));
        }
        ((v4bf*)hi)[i] = h;
        ((v4bf*)lo)[i] = l;
    }
}

// ---------------------------------------------------------------------------
// LIF layer-1: mem1 = beta*mem1 + cur1 - (mem1>1);  spk1 = (mem1_new>1) as bf16
// ---------------------------------------------------------------------------
__global__ void snn_lif1_kernel(const float* __restrict__ cur1,
                                float* __restrict__ mem1,
                                __bf16* __restrict__ spk1,
                                float beta, int n4) {
    int i = blockIdx.x * blockDim.x + threadIdx.x;
    if (i < n4) {
        const __bf16 one  = __builtin_bit_cast(__bf16, (unsigned short)0x3F80);
        const __bf16 zero = __builtin_bit_cast(__bf16, (unsigned short)0x0000);
        v4f c = ((const v4f*)cur1)[i];
        v4f m = ((const v4f*)mem1)[i];
        v4f mn; v4bf sp;
#pragma unroll
        for (int j = 0; j < 4; j++) {
            float rst = (m[j] > 1.0f) ? 1.0f : 0.0f;
            float x   = beta * m[j] + c[j] - rst;
            mn[j] = x;
            sp[j] = (x > 1.0f) ? one : zero;
        }
        ((v4f*)mem1)[i]  = mn;
        ((v4bf*)spk1)[i] = sp;
    }
}

// ---------------------------------------------------------------------------
// Fragment loaders from LDS (K-contiguous tiles, row stride STR bf16).
// A (16x32): lanes 0-15 hold K runs [0..7],[16..23]; lanes 16-31 [8..15],[24..31].
// B (32x16): same K runs as A: lanes 0-15 [0..7],[16..23]; lanes 16-31 [8..15],[24..31].
// ---------------------------------------------------------------------------
__device__ inline v16bf frag_a(const __bf16* rowbase, int g) {
    v8bf p0 = *(const v8bf*)(rowbase + 8 * g);
    v8bf p1 = *(const v8bf*)(rowbase + 16 + 8 * g);
    v16bf r;
#pragma unroll
    for (int i = 0; i < 8; i++) { r[i] = p0[i]; r[i + 8] = p1[i]; }
    return r;
}
__device__ inline v16bf frag_b(const __bf16* rowbase, int g) {
    v8bf p0 = *(const v8bf*)(rowbase + 8 * g);
    v8bf p1 = *(const v8bf*)(rowbase + 16 + 8 * g);
    v16bf r;
#pragma unroll
    for (int i = 0; i < 8; i++) { r[i] = p0[i]; r[i + 8] = p1[i]; }
    return r;
}

// Issue async loads for one R x KC bf16 tile (K-contiguous source).
// Each thread issues R/32 x b128 (16B) transfers.
template <int R>
__device__ inline void stage_async(const __bf16* __restrict__ src, int row0, int K,
                                   int k0, unsigned lds_base_bytes, int tid) {
#pragma unroll
    for (int j = 0; j < R / 32; ++j) {
        const int seg = tid + 256 * j;
        const int row = seg >> 3;            // 8 segments (of 8 bf16) per row
        const int col = (seg & 7) * 8;
        const __bf16* g = src + (size_t)(row0 + row) * (size_t)K + k0 + col;
        async_b128(lds_base_bytes + (unsigned)(row * STR + col) * 2u, g);
    }
}

// ---------------------------------------------------------------------------
// Split-bf16 WMMA GEMM:  C[M,N] = A[M,K] @ Bt[N,K]^T + bias[N]
// MODE 0:  A split (Ahi,Alo): AhBh + AhBl + AlBh; store C -> Cout
// MODE 1:  A exact bf16 (spk1): ABh + ABl; fused LIF2 epilogue
// 256 threads = 8 wave32; waves tiled (TM/32) x (8/(TM/32)); per-wave 32 x (TN/WCOLS).
// Async global->LDS staging, double-buffered; V_WMMA_F32_16X16X32_BF16 core.
// ---------------------------------------------------------------------------
template <int MODE, int TM, int TN>
__global__ __launch_bounds__(256)
void snn_gemm_bf16(const __bf16* __restrict__ Ahi, const __bf16* __restrict__ Alo,
                   const __bf16* __restrict__ Bhi, const __bf16* __restrict__ Blo,
                   const float* __restrict__ bias,
                   float* __restrict__ Cout,
                   float* __restrict__ mem2,
                   float* __restrict__ spk_out,
                   float* __restrict__ mem_out,
                   int M, int N, int K, float beta) {
    constexpr int WROWS = TM / 32;           // waves along M
    constexpr int WCOLS = 8 / WROWS;         // waves along N
    constexpr int FM = 2;                    // 16-row fragments per wave
    constexpr int FN = TN / WCOLS / 16;      // 16-col fragments per wave
    constexpr int TSA = TM * STR;            // bf16 elems per A tile
    constexpr int TSB = TN * STR;
    constexpr int ABOFF = ((MODE == 0) ? 2 : 1) * TSA;   // B tiles offset in buffer
    constexpr int BUF = ABOFF + 2 * TSB;                 // bf16 elems per buffer
    constexpr int INFLIGHT = ((MODE == 0) ? 2 * TM / 32 : TM / 32) + 2 * (TN / 32);

    __shared__ __bf16 smem[2 * BUF];

    const int tid  = threadIdx.x;
    const int lane = tid & 31;
    const int wave = tid >> 5;
    const int wm   = wave % WROWS;
    const int wn   = wave / WROWS;
    const int lrow = lane & 15;
    const int g    = lane >> 4;

    const int tileM = blockIdx.y * TM;
    const int tileN = blockIdx.x * TN;

    const unsigned sbase = (unsigned)(uintptr_t)(void*)smem;  // LDS byte offset (addr[31:0])

    v8f acc[FM][FN] = {};

    auto issue = [&](int k0, int buf) {
        const unsigned b = sbase + (unsigned)(buf * BUF) * 2u;
        stage_async<TM>(Ahi, tileM, K, k0, b, tid);
        if (MODE == 0) stage_async<TM>(Alo, tileM, K, k0, b + (unsigned)TSA * 2u, tid);
        stage_async<TN>(Bhi, tileN, K, k0, b + (unsigned)ABOFF * 2u, tid);
        stage_async<TN>(Blo, tileN, K, k0, b + (unsigned)(ABOFF + TSB) * 2u, tid);
    };

    const int nchunk = K / KC;
    issue(0, 0);

    for (int c = 0; c < nchunk; ++c) {
        if (c + 1 < nchunk) {
            issue((c + 1) * KC, (c + 1) & 1);
            wait_async_le<INFLIGHT>();   // in-order completion: drains chunk c
        } else {
            wait_async_le<0>();
        }
        __syncthreads();                 // chunk c visible to all waves

        const __bf16* base = smem + (c & 1) * BUF;
        const __bf16* As_hi = base;
        const __bf16* As_lo = base + TSA;        // MODE 0 only
        const __bf16* Bs_hi = base + ABOFF;
        const __bf16* Bs_lo = Bs_hi + TSB;

#pragma unroll
        for (int kk = 0; kk < KC; kk += 32) {
            v16bf ah[FM], al[FM], bh[FN], bl[FN];
#pragma unroll
            for (int fm = 0; fm < FM; fm++) {
                const int row = (wm * 32 + fm * 16 + lrow) * STR + kk;
                ah[fm] = frag_a(&As_hi[row], g);
                if (MODE == 0) al[fm] = frag_a(&As_lo[row], g);
            }
#pragma unroll
            for (int fn = 0; fn < FN; fn++) {
                const int row = (wn * (FN * 16) + fn * 16 + lrow) * STR + kk;
                bh[fn] = frag_b(&Bs_hi[row], g);
                bl[fn] = frag_b(&Bs_lo[row], g);
            }
#pragma unroll
            for (int fm = 0; fm < FM; fm++) {
#pragma unroll
                for (int fn = 0; fn < FN; fn++) {
                    acc[fm][fn] = __builtin_amdgcn_wmma_f32_16x16x32_bf16(
                        false, ah[fm], false, bh[fn], (short)0, acc[fm][fn], false, false);
                    acc[fm][fn] = __builtin_amdgcn_wmma_f32_16x16x32_bf16(
                        false, ah[fm], false, bl[fn], (short)0, acc[fm][fn], false, false);
                    if (MODE == 0)
                        acc[fm][fn] = __builtin_amdgcn_wmma_f32_16x16x32_bf16(
                            false, al[fm], false, bh[fn], (short)0, acc[fm][fn], false, false);
                }
            }
        }
        __syncthreads();   // all waves done reading buffer c before it is re-staged
    }

    // Epilogue. C/D layout: VGPR i -> M = i (+8 for lanes 16-31), N = lane & 15.
    const int mhalf = g << 3;
#pragma unroll
    for (int fm = 0; fm < FM; fm++) {
#pragma unroll
        for (int fn = 0; fn < FN; fn++) {
            const int n  = tileN + wn * (FN * 16) + fn * 16 + lrow;
            const float bi = bias[n];
#pragma unroll
            for (int i = 0; i < 8; i++) {
                const int m = tileM + wm * 32 + fm * 16 + mhalf + i;
                const size_t idx = (size_t)m * (size_t)N + (size_t)n;
                const float cur = acc[fm][fn][i] + bi;
                if (MODE == 0) {
                    Cout[idx] = cur;
                } else {
                    float mprev = mem2[idx];
                    float rst   = (mprev > 1.0f) ? 1.0f : 0.0f;
                    float mnew  = beta * mprev + cur - rst;
                    float spk   = (mnew > 1.0f) ? 1.0f : 0.0f;
                    mem2[idx]    = mnew;
                    spk_out[idx] = spk;
                    mem_out[idx] = mnew;
                }
            }
        }
    }
}

// ---------------------------------------------------------------------------
// Launch
// ---------------------------------------------------------------------------
extern "C" void kernel_launch(void* const* d_in, const int* in_sizes, int n_in,
                              void* d_out, int out_size, void* d_ws, size_t ws_size,
                              hipStream_t stream) {
    (void)in_sizes; (void)n_in; (void)out_size; (void)ws_size;

    const float* x  = (const float*)d_in[0];   // [1024, 2048]
    const float* W1 = (const float*)d_in[1];   // [4096, 2048]
    const float* b1 = (const float*)d_in[2];   // [4096]
    const float* W2 = (const float*)d_in[3];   // [1024, 4096]
    const float* b2 = (const float*)d_in[4];   // [1024]
    float* out = (float*)d_out;                // spk2_rec[25,1024,1024] ++ mem2_rec[...]

    const int B = 1024, NIN = 2048, NH = 4096, NO = 1024, T = 25;
    const float beta = 0.95f;

    // Workspace layout
    char* w = (char*)d_ws;
    float*  cur1 = (float*)w;            w += (size_t)B * NH * 4;
    float*  mem1 = (float*)w;            w += (size_t)B * NH * 4;
    float*  mem2 = (float*)w;            w += (size_t)B * NO * 4;
    __bf16* spk1 = (__bf16*)w;           w += (size_t)B * NH * 2;
    __bf16* xhi  = (__bf16*)w;           w += (size_t)B * NIN * 2;
    __bf16* xlo  = (__bf16*)w;           w += (size_t)B * NIN * 2;
    __bf16* W1hi = (__bf16*)w;           w += (size_t)NH * NIN * 2;
    __bf16* W1lo = (__bf16*)w;           w += (size_t)NH * NIN * 2;
    __bf16* W2hi = (__bf16*)w;           w += (size_t)NO * NH * 2;
    __bf16* W2lo = (__bf16*)w;           w += (size_t)NO * NH * 2;

    // One-time operand splits
    snn_split_kernel<<<(B * NIN / 4) / 256, 256, 0, stream>>>(x,  xhi,  xlo,  B * NIN / 4);
    snn_split_kernel<<<(NH * NIN / 4) / 256, 256, 0, stream>>>(W1, W1hi, W1lo, NH * NIN / 4);
    snn_split_kernel<<<(NO * NH / 4) / 256, 256, 0, stream>>>(W2, W2hi, W2lo, NO * NH / 4);

    // Zero membrane state
    snn_zero_kernel<<<(B * NH / 4) / 256, 256, 0, stream>>>(mem1, B * NH / 4);
    snn_zero_kernel<<<(B * NO / 4) / 256, 256, 0, stream>>>(mem2, B * NO / 4);

    // cur1 = x @ W1^T + b1  (bf16x3 split; 128x128 tiles, 256 WGs)
    snn_gemm_bf16<0, 128, 128><<<dim3(NH / 128, B / 128), 256, 0, stream>>>(
        xhi, xlo, W1hi, W1lo, b1, cur1, nullptr, nullptr, nullptr, B, NH, NIN, beta);

    for (int t = 0; t < T; ++t) {
        snn_lif1_kernel<<<(B * NH / 4) / 256, 256, 0, stream>>>(cur1, mem1, spk1, beta, B * NH / 4);
        // cur2 = spk1 @ W2^T + b2; 64x128 tiles -> 128 WGs to fill the device
        snn_gemm_bf16<1, 64, 128><<<dim3(NO / 128, B / 64), 256, 0, stream>>>(
            spk1, nullptr, W2hi, W2lo, b2, nullptr, mem2,
            out + (size_t)t * B * NO,
            out + (size_t)(T + t) * B * NO,
            B, NO, NH, beta);
    }
}
